// DJconv_45724221833875
// MI455X (gfx1250) — matrix operand order repaired and checked by reading, not compiled, not run on hardware
//
#include <hip/hip_runtime.h>

typedef __attribute__((ext_vector_type(16))) _Float16 v16h;
typedef __attribute__((ext_vector_type(8)))  float    v8f;

#define NN   131072
#define EE   256
#define INC  128
#define OUTC 256

// ---------------------------------------------------------------------------
// Kernel 0: zero the global gram-nonzero bitset (256 rows x 8 dwords)
// ---------------------------------------------------------------------------
__global__ void k_init_g(unsigned* __restrict__ G) {
    for (int i = threadIdx.x; i < 2048; i += 256) G[i] = 0u;
}

// ---------------------------------------------------------------------------
// Kernel 1: build per-node 256-bit masks R_n = (H[n,:] >= 0.5), and
// accumulate G'[k] |= R_n for every k set in R_n (gram>0 pattern).
// One wave handles one node at a time: ballot over 32 lanes x 8 rounds gives
// fully coalesced 128B reads of H. LDS bitmap per workgroup, merged with
// global atomicOr at the end.
// grid: 256 blocks x 256 threads, 512 nodes / block, 64 / wave.
// ---------------------------------------------------------------------------
__global__ __launch_bounds__(256) void k_masks(const float* __restrict__ H,
                                               unsigned* __restrict__ masks,
                                               unsigned* __restrict__ G) {
    __shared__ unsigned Gl[2048];
    const int tid  = threadIdx.x;
    const int lane = tid & 31;
    const int wave = tid >> 5;
    for (int i = tid; i < 2048; i += 256) Gl[i] = 0u;
    __syncthreads();

    const int n0 = blockIdx.x * 512 + wave * 64;
    for (int i = 0; i < 64; ++i) {
        const int n = n0 + i;
        const float* row = H + (size_t)n * EE;
        unsigned mym = 0u;          // lane r (<8) ends up holding mask dword r
        unsigned mall[8];           // wave-uniform copies of all 8 dwords
#pragma unroll
        for (int r = 0; r < 8; ++r) {
            const float v = row[r * 32 + lane];
            const unsigned b = __builtin_amdgcn_ballot_w32(v >= 0.5f);
            mall[r] = b;
            if (lane == r) mym = b;
        }
        if (lane < 8) masks[(size_t)n * 8 + lane] = mym;
        // G'[k] |= R_n for each set bit k (uniform loop, ~12.8 bits/node)
#pragma unroll
        for (int r = 0; r < 8; ++r) {
            unsigned m = mall[r];
            while (m) {
                const int k = r * 32 + __builtin_ctz(m);
                m &= m - 1u;
                if (lane < 8) atomicOr(&Gl[k * 8 + lane], mym);
            }
        }
    }
    __syncthreads();
    for (int i = tid; i < 2048; i += 256) {
        const unsigned v = Gl[i];
        if (v) atomicOr(&G[i], v);
    }
}

// ---------------------------------------------------------------------------
// Kernel 2: dv[n] = popc(R_n) + popc(OR_{k in R_n} G'[k]);  s[n] = 1 + dv^-1/2
// G' (8KB) staged in LDS; per-thread b128 mask loads + b128 LDS gathers.
// grid: 512 x 256 (one thread per node)
// ---------------------------------------------------------------------------
__global__ __launch_bounds__(256) void k_scale(const unsigned* __restrict__ masks,
                                               const unsigned* __restrict__ G,
                                               float* __restrict__ s) {
    __shared__ __align__(16) unsigned Gs[2048];
    const int tid = threadIdx.x;
    for (int i = tid; i < 2048; i += 256) Gs[i] = G[i];
    __syncthreads();

    const int n = blockIdx.x * 256 + tid;
    const uint4 m0 = *(const uint4*)(masks + (size_t)n * 8);
    const uint4 m1 = *(const uint4*)(masks + (size_t)n * 8 + 4);
    const unsigned md[8] = {m0.x, m0.y, m0.z, m0.w, m1.x, m1.y, m1.z, m1.w};

    int dv = 0;
    uint4 a0 = {0u, 0u, 0u, 0u};
    uint4 a1 = {0u, 0u, 0u, 0u};
#pragma unroll
    for (int d = 0; d < 8; ++d) {
        unsigned m = md[d];
        dv += __builtin_popcount(m);
        while (m) {
            const int k = d * 32 + __builtin_ctz(m);
            m &= m - 1u;
            const uint4* gr = (const uint4*)(&Gs[k * 8]);
            const uint4 g0 = gr[0], g1 = gr[1];
            a0.x |= g0.x; a0.y |= g0.y; a0.z |= g0.z; a0.w |= g0.w;
            a1.x |= g1.x; a1.y |= g1.y; a1.z |= g1.z; a1.w |= g1.w;
        }
    }
    dv += __builtin_popcount(a0.x) + __builtin_popcount(a0.y) +
          __builtin_popcount(a0.z) + __builtin_popcount(a0.w) +
          __builtin_popcount(a1.x) + __builtin_popcount(a1.y) +
          __builtin_popcount(a1.z) + __builtin_popcount(a1.w);

    s[n] = 1.0f + (dv > 0 ? __frsqrt_rn((float)dv) : 0.0f);
}

// ---------------------------------------------------------------------------
// Kernel 3: split fp32 W[k,n] into hi/lo f16 and transpose to [n,k] so the
// WMMA B-fragment (16 consecutive K per lane) is one contiguous 32B load.
// grid: 128 x 256
// ---------------------------------------------------------------------------
__global__ void k_splitW(const float* __restrict__ W,
                         _Float16* __restrict__ WhiT,
                         _Float16* __restrict__ WloT) {
    const int idx = blockIdx.x * 256 + threadIdx.x;   // idx = k*256 + n
    const int k = idx >> 8;
    const int n = idx & 255;
    const float w  = W[idx];
    const _Float16 hi = (_Float16)w;
    const _Float16 lo = (_Float16)(w - (float)hi);
    WhiT[n * INC + k] = hi;
    WloT[n * INC + k] = lo;
}

// ---------------------------------------------------------------------------
// Kernel 4: out = diag(s) * (U @ W) + bias via split-f16 WMMA (hi*hi+hi*lo+lo*hi,
// f32 accumulate). Block = 8 waves; wave tile = 16(M) x 64(N); block = 128x64.
// K = 128 -> 4 k-steps x 4 col-tiles x 3 wmma = 48 v_wmma per wave.
// grid: dim3(OUTC/64, NN/128) = (4, 1024)
// ---------------------------------------------------------------------------
__global__ __launch_bounds__(256) void k_gemm(const float* __restrict__ U,
                                              const _Float16* __restrict__ WhiT,
                                              const _Float16* __restrict__ WloT,
                                              const float* __restrict__ s,
                                              const float* __restrict__ bias,
                                              float* __restrict__ out) {
    const int lane = threadIdx.x & 31;
    const int wave = threadIdx.x >> 5;
    const int r0   = blockIdx.y * 128 + wave * 16;   // tile row base
    const int c0   = blockIdx.x * 64;                // tile col base
    const int M    = lane & 15;
    const int half = lane >> 4;                      // lane half (0/1)
    const int ca   = half * 16;                      // A: K offsets {ca..ca+15}
    const int kb   = half * 16;                      // B: K offsets {kb..kb+15}

    v8f acc[4];
#pragma unroll
    for (int t = 0; t < 4; ++t) acc[t] = (v8f){0.f, 0.f, 0.f, 0.f, 0.f, 0.f, 0.f, 0.f};

    const float* urow = U + (size_t)(r0 + M) * INC;

#pragma unroll
    for (int ks = 0; ks < 4; ++ks) {
        const int K0 = ks * 32;
        const float* up = urow + K0 + ca;
        const float4 f0a = *(const float4*)(up);
        const float4 f0b = *(const float4*)(up + 4);
        const float4 f1a = *(const float4*)(up + 8);
        const float4 f1b = *(const float4*)(up + 12);
        const float f[16] = {f0a.x, f0a.y, f0a.z, f0a.w, f0b.x, f0b.y, f0b.z, f0b.w,
                             f1a.x, f1a.y, f1a.z, f1a.w, f1b.x, f1b.y, f1b.z, f1b.w};
        v16h ahi, alo;
#pragma unroll
        for (int j = 0; j < 16; ++j) {
            const _Float16 h = (_Float16)f[j];
            ahi[j] = h;
            alo[j] = (_Float16)(f[j] - (float)h);
        }

        const size_t wb = (size_t)(c0 + M) * INC + (size_t)(K0 + kb);
#pragma unroll
        for (int t = 0; t < 4; ++t) {
            const size_t o = wb + (size_t)(16 * t) * INC;
            const v16h bhi = *(const v16h*)(WhiT + o);
            const v16h blo = *(const v16h*)(WloT + o);
            acc[t] = __builtin_amdgcn_wmma_f32_16x16x32_f16(false, ahi, false, bhi,
                                                            (short)0, acc[t], false, false);
            acc[t] = __builtin_amdgcn_wmma_f32_16x16x32_f16(false, ahi, false, blo,
                                                            (short)0, acc[t], false, false);
            acc[t] = __builtin_amdgcn_wmma_f32_16x16x32_f16(false, alo, false, bhi,
                                                            (short)0, acc[t], false, false);
        }
    }

    // Epilogue: C layout -> lane col = lane%16, VGPR r -> row = half*8 + r
    float sv[8];
#pragma unroll
    for (int r = 0; r < 8; ++r) sv[r] = s[r0 + half * 8 + r];
#pragma unroll
    for (int t = 0; t < 4; ++t) {
        const int col = c0 + 16 * t + M;
        const float bv = bias[col];
#pragma unroll
        for (int r = 0; r < 8; ++r) {
            const int row = r0 + half * 8 + r;
            out[(size_t)row * OUTC + col] = sv[r] * acc[t][r] + bv;
        }
    }
}

// ---------------------------------------------------------------------------
extern "C" void kernel_launch(void* const* d_in, const int* in_sizes, int n_in,
                              void* d_out, int out_size, void* d_ws, size_t ws_size,
                              hipStream_t stream) {
    (void)in_sizes; (void)n_in; (void)out_size; (void)ws_size;
    const float* H    = (const float*)d_in[0];   // [N, E]
    const float* U    = (const float*)d_in[1];   // [N, 128]
    const float* W    = (const float*)d_in[2];   // [128, 256]
    const float* bias = (const float*)d_in[3];   // [256]
    float* out = (float*)d_out;                  // [N, 256]

    char* ws = (char*)d_ws;
    unsigned*  masks = (unsigned*)ws;                          // 4 MB
    unsigned*  G     = (unsigned*)(ws + (4u << 20));           // 8 KB
    float*     s     = (float*)(ws + (4u << 20) + (64u << 10)); // 512 KB
    _Float16*  WhiT  = (_Float16*)(ws + (5u << 20));           // 64 KB
    _Float16*  WloT  = (_Float16*)(ws + (5u << 20) + (64u << 10)); // 64 KB

    k_init_g<<<1, 256, 0, stream>>>(G);
    k_masks<<<256, 256, 0, stream>>>(H, masks, G);
    k_scale<<<512, 256, 0, stream>>>(masks, G, s);
    k_splitW<<<128, 256, 0, stream>>>(W, WhiT, WloT);
    k_gemm<<<dim3(OUTC / 64, NN / 128), 256, 0, stream>>>(U, WhiT, WloT, s, bias, out);
}
